// CMT_54752243089522
// MI455X (gfx1250) — matrix operand order repaired and checked by reading, not compiled, not run on hardware
//
#include <hip/hip_runtime.h>
#include <hip/hip_bf16.h>

typedef __attribute__((ext_vector_type(16))) _Float16 v16h;
typedef __attribute__((ext_vector_type(8)))  float    v8f;

#define WMMA_F16(a,b,c) __builtin_amdgcn_wmma_f32_16x16x32_f16(false,(a),false,(b),(short)0,(c),false,false)

__device__ __forceinline__ v8f zero8() {
  v8f z;
#pragma unroll
  for (int i = 0; i < 8; ++i) z[i] = 0.0f;
  return z;
}

// CDNA5 async global->LDS copy (16B per lane), tracked by ASYNCcnt.
// VDST operand = wave-local LDS byte offset == low 32 bits of the flat shared address
// (ISA 10.2: LDS aperture keeps the offset in addr[31:0]).
__device__ __forceinline__ void async_copy_b128(_Float16* lds_dst, const _Float16* gsrc) {
  unsigned loff = (unsigned)(uintptr_t)lds_dst;
  asm volatile("global_load_async_to_lds_b128 %0, %1, off"
               :: "v"(loff), "v"(gsrc)
               : "memory");
}
__device__ __forceinline__ void wait_async0() {
  asm volatile("s_wait_asynccnt 0" ::: "memory");
}

// A-fragment (16x32 f16): lanes 0-15: M=lane, K = {0..7, 16..23}; lanes 16-31: M=lane-16, K = {8..15, 24..31}
__device__ __forceinline__ v16h ld_fragA(const _Float16* p0, int ld) {
  const int lane = threadIdx.x & 31;
  const _Float16* p = p0 + (lane & 15) * ld + ((lane >> 4) << 3);
  v16h f;
#pragma unroll
  for (int i = 0; i < 8; ++i) f[i] = p[i];
#pragma unroll
  for (int i = 0; i < 8; ++i) f[i + 8] = p[i + 16];
  return f;
}

// B-fragment (32x16 f16) from an N-major (transposed) tile: lane n = lane&15,
// lanes 0-15: K = {0..7, 16..23}, lanes 16-31: K = {8..15, 24..31} (per ISA B-matrix table pattern)
__device__ __forceinline__ v16h ld_fragB(const _Float16* p0, int ld) {
  const int lane = threadIdx.x & 31;
  const _Float16* p = p0 + (lane & 15) * ld + ((lane >> 4) << 3);
  v16h f;
#pragma unroll
  for (int i = 0; i < 8; ++i) f[i] = p[i];
#pragma unroll
  for (int i = 0; i < 8; ++i) f[i + 8] = p[i + 16];
  return f;
}

// ---------------------------------------------------------------------------
// Generic tiled GEMM:  out[M x N] = act(A_f16[M x K] @ W_f32[K x N] + bias) (+ resid)
// act: 0 = none, 1 = exact GELU, 2 = elu(x)+1
// ---------------------------------------------------------------------------
#define BM 128
#define BN 128
#define BKK 32
#define LDT 40

__global__ __launch_bounds__(256) void gemm_kernel(
    const _Float16* __restrict__ A, int lda,
    const float* __restrict__ W, const float* __restrict__ bias,
    const float* __restrict__ resid,
    float* __restrict__ out32, _Float16* __restrict__ out16,
    int ldo, int nc0, int M, int N, int K, int act)
{
  __shared__ _Float16 sA[BM * LDT];
  __shared__ _Float16 sB[BN * LDT];   // transposed weight tile: [n][k]
  const int row0 = blockIdx.y * BM;
  const int col0 = blockIdx.x * BN;
  const int tid = threadIdx.x;
  const int wid = tid >> 5, lane = tid & 31;
  const int wr = wid >> 1, wc = wid & 1;  // 4x2 wave grid -> each wave 32 rows x 64 cols
  const bool interior = (row0 + BM <= M); // block-uniform
  v8f acc[2][4];
#pragma unroll
  for (int i = 0; i < 2; ++i)
#pragma unroll
    for (int j = 0; j < 4; ++j) acc[i][j] = zero8();

  for (int k0 = 0; k0 < K; k0 += BKK) {
    if (k0 + BKK < K) __builtin_prefetch(W + (size_t)(k0 + BKK) * N + col0, 0, 1);
    // stage A tile 128x32 f16
    if (interior) {
      for (int i = tid; i < BM * 4; i += 256) {      // async DMA path, no VGPR round-trip
        int r = i >> 2, c = (i & 3) << 3;
        async_copy_b128(sA + r * LDT + c, A + (size_t)(row0 + r) * lda + (k0 + c));
      }
    } else {
      for (int i = tid; i < BM * 4; i += 256) {      // guarded tail path
        int r = i >> 2, c = (i & 3) << 3;
        uint4 v = {0u, 0u, 0u, 0u};
        int gr = row0 + r;
        if (gr < M) v = *(const uint4*)(A + (size_t)gr * lda + (k0 + c));
        *(uint4*)(sA + r * LDT + c) = v;
      }
    }
    // stage W tile 32x128 f32 -> f16 transposed [n][k] (needs VALU convert)
    for (int i = tid; i < BKK * BN; i += 256) {
      int n = i & (BN - 1), k = i >> 7;
      int gn = col0 + n;
      float w = (gn < N) ? W[(size_t)(k0 + k) * N + gn] : 0.0f;
      sB[n * LDT + k] = (_Float16)w;
    }
    wait_async0();
    __syncthreads();
    v16h af0 = ld_fragA(sA + (wr * 32) * LDT, LDT);
    v16h af1 = ld_fragA(sA + (wr * 32 + 16) * LDT, LDT);
#pragma unroll
    for (int j = 0; j < 4; ++j) {
      v16h bf = ld_fragB(sB + (wc * 64 + j * 16) * LDT, LDT);
      acc[0][j] = WMMA_F16(af0, bf, acc[0][j]);
      acc[1][j] = WMMA_F16(af1, bf, acc[1][j]);
    }
    __syncthreads();
  }

  const int mb = (lane >> 4) << 3, nn = lane & 15;
#pragma unroll
  for (int i = 0; i < 2; ++i)
#pragma unroll
    for (int j = 0; j < 4; ++j) {
      int gn = col0 + wc * 64 + j * 16 + nn;
      if (gn >= N) continue;
      float bv = bias ? bias[gn] : 0.0f;
      int gm0 = row0 + wr * 32 + i * 16 + mb;
#pragma unroll
      for (int r = 0; r < 8; ++r) {
        int gm = gm0 + r;
        if (gm >= M) break;
        float vv = acc[i][j][r] + bv;
        if (act == 1)      vv = 0.5f * vv * (1.0f + erff(vv * 0.70710678118654752f));
        else if (act == 2) vv = (vv > 0.0f) ? (vv + 1.0f) : __expf(vv);
        size_t oidx = (size_t)gm * ldo + nc0 + gn;
        if (resid) vv += resid[oidx];
        if (out32) out32[oidx] = vv;
        if (out16) out16[oidx] = (_Float16)vv;
      }
    }
}

// ---------------------------------------------------------------------------
// Causal linear attention, chunked scan. One workgroup per (batch, head).
// Q,K already hold fq=elu(q)+1 / fk=elu(k)+1 in f16.
// ---------------------------------------------------------------------------
#define CH 128
#define DH 64
#define LQ 72
#define LT 136

__global__ __launch_bounds__(256) void attn_kernel(
    const _Float16* __restrict__ Q, const _Float16* __restrict__ Kf,
    const _Float16* __restrict__ Vf, _Float16* __restrict__ Aout)
{
  __shared__ _Float16 sq[CH * LQ];     // fq chunk [t][d]
  __shared__ _Float16 sk[CH * LQ];     // fk chunk [t][d]   (B operand for scores)
  __shared__ _Float16 skT[DH * LT];    // fk^T [d][t]       (A operand for state update)
  __shared__ _Float16 svT[DH * LT];    // v^T  [d][t]       (B operand, N-major)
  __shared__ _Float16 ssc[CH * LT];    // masked scores f16 [t][t']
  __shared__ _Float16 sST[DH * LQ];    // S^T snapshot f16  [e][d]
  __shared__ float Sst[DH * DH];       // running state S[d][e], f32
  __shared__ float zst[DH];
  __shared__ float den[CH];

  const int tid = threadIdx.x, wid = tid >> 5, lane = tid & 31;
  const int b = blockIdx.x >> 3, hh = blockIdx.x & 7;
  const size_t baserow = (size_t)b * 2048;
  const int colq = hh * DH;
  const int mb = (lane >> 4) << 3, nn = lane & 15;
  const int rb = wid * 16;   // each wave owns a 16-row band of the chunk

  for (int i = tid; i < DH * DH; i += 256) Sst[i] = 0.0f;
  if (tid < DH) zst[tid] = 0.0f;
  __syncthreads();

  for (int c = 0; c < 16; ++c) {
    const size_t row0 = baserow + (size_t)c * CH;
    // phase 1a: async-load fq, fk rows; v transposed; snapshot S -> f16 transposed
    for (int i = tid; i < CH * 8; i += 256) {
      int t = i >> 3, d = (i & 7) << 3;
      async_copy_b128(sq + t * LQ + d, Q  + (row0 + t) * 512 + colq + d);
      async_copy_b128(sk + t * LQ + d, Kf + (row0 + t) * 512 + colq + d);
    }
    for (int i = tid; i < CH * DH; i += 256) {
      int t = i >> 6, d = i & 63;
      svT[d * LT + t] = Vf[(row0 + t) * 512 + colq + d];
    }
    for (int i = tid; i < DH * DH; i += 256) {
      int dd = i >> 6, e = i & 63;
      sST[e * LQ + dd] = (_Float16)Sst[dd * DH + e];
    }
    wait_async0();
    __syncthreads();
    // phase 1b: fk transposed (LDS -> LDS)
    for (int i = tid; i < CH * DH; i += 256) {
      int t = i >> 6, d = i & 63;
      skT[d * LT + t] = sk[t * LQ + d];
    }
    __syncthreads();

    // phase 2: scores sc = fq @ fk^T for own band (K = 64), mask, rowsum, store f16
    v8f sacc[8];
#pragma unroll
    for (int j = 0; j < 8; ++j) sacc[j] = zero8();
    {
      v16h a0 = ld_fragA(sq + rb * LQ, LQ);
      v16h a1 = ld_fragA(sq + rb * LQ + 32, LQ);
#pragma unroll
      for (int j = 0; j < 8; ++j) {
        v16h b0 = ld_fragB(sk + (j * 16) * LQ, LQ);
        v16h b1 = ld_fragB(sk + (j * 16) * LQ + 32, LQ);
        sacc[j] = WMMA_F16(a0, b0, sacc[j]);
        sacc[j] = WMMA_F16(a1, b1, sacc[j]);
      }
    }
    float rs[8];
#pragma unroll
    for (int r = 0; r < 8; ++r) rs[r] = 0.0f;
#pragma unroll
    for (int j = 0; j < 8; ++j) {
      int ncol = j * 16 + nn;
#pragma unroll
      for (int r = 0; r < 8; ++r) {
        int mrow = rb + mb + r;
        float sv = sacc[j][r];
        if (ncol > mrow) sv = 0.0f;        // causal (tril) mask inside chunk
        rs[r] += sv;
        ssc[mrow * LT + ncol] = (_Float16)sv;
      }
    }
#pragma unroll
    for (int r = 0; r < 8; ++r) {          // reduce across the 16 lanes of each half-wave
      float s = rs[r];
      s += __shfl_xor(s, 1);
      s += __shfl_xor(s, 2);
      s += __shfl_xor(s, 4);
      s += __shfl_xor(s, 8);
      if (nn == 0) den[rb + mb + r] = s;
    }
    __syncthreads();

    // phase 3: den += fq . z  (+ eps)
    if (tid < CH) {
      float s = 0.0f;
      for (int d = 0; d < DH; ++d) s += (float)sq[tid * LQ + d] * zst[d];
      den[tid] += s + 1e-6f;
    }
    __syncthreads();

    // phase 4: num = sc @ v (K=128) + fq @ S (K=64); out = num / den
    v8f oacc[4];
#pragma unroll
    for (int j = 0; j < 4; ++j) oacc[j] = zero8();
#pragma unroll
    for (int ks = 0; ks < 4; ++ks) {
      v16h a = ld_fragA(ssc + rb * LT + ks * 32, LT);
#pragma unroll
      for (int j = 0; j < 4; ++j)
        oacc[j] = WMMA_F16(a, ld_fragB(svT + (j * 16) * LT + ks * 32, LT), oacc[j]);
    }
#pragma unroll
    for (int ks = 0; ks < 2; ++ks) {
      v16h a = ld_fragA(sq + rb * LQ + ks * 32, LQ);
#pragma unroll
      for (int j = 0; j < 4; ++j)
        oacc[j] = WMMA_F16(a, ld_fragB(sST + (j * 16) * LQ + ks * 32, LQ), oacc[j]);
    }
#pragma unroll
    for (int j = 0; j < 4; ++j) {
      int gn = colq + j * 16 + nn;
#pragma unroll
      for (int r = 0; r < 8; ++r) {
        int mrow = rb + mb + r;
        float o = oacc[j][r] / den[mrow];
        Aout[(row0 + mrow) * 512 + gn] = (_Float16)o;
      }
    }

    // phase 5: S += fk^T @ v (each wave 2 of 16 tiles); z += colsum(fk)
#pragma unroll
    for (int tt = 0; tt < 2; ++tt) {
      int tile = wid * 2 + tt;
      int mt = (tile >> 2) * 16, nt = (tile & 3) * 16;
      v8f u = zero8();
#pragma unroll
      for (int ks = 0; ks < 4; ++ks)
        u = WMMA_F16(ld_fragA(skT + mt * LT + ks * 32, LT),
                     ld_fragB(svT + nt * LT + ks * 32, LT), u);
#pragma unroll
      for (int r = 0; r < 8; ++r)
        Sst[(mt + mb + r) * DH + nt + nn] += u[r];
    }
    if (tid < DH) {
      float s = 0.0f;
      for (int t = 0; t < CH; ++t) s += (float)skT[tid * LT + t];
      zst[tid] += s;
    }
    __syncthreads();
  }
}

// ---------------------------------------------------------------------------
// Embedding gather: E[8192 x 896] f16 (scaled concat embeddings), T[8192 x 256] f16
// ---------------------------------------------------------------------------
__global__ void embed_gather_kernel(
    const int* __restrict__ x,
    const float* __restrict__ e0, const float* __restrict__ e1,
    const float* __restrict__ e2, const float* __restrict__ e3,
    const float* __restrict__ e4, const float* __restrict__ e5,
    const float* __restrict__ e6, const float* __restrict__ etime,
    _Float16* __restrict__ E, _Float16* __restrict__ T)
{
  int r = blockIdx.x;
  int b = r >> 11, tt = r & 2047;
  _Float16* Er = E + (size_t)r * 896;
  _Float16* Tr = T + (size_t)r * 256;
  if (tt < 7) {
    for (int c = threadIdx.x; c < 896; c += blockDim.x) Er[c] = (_Float16)0.0f;
    for (int c = threadIdx.x; c < 256; c += blockDim.x) Tr[c] = (_Float16)0.0f;
    return;
  }
  int t = tt - 7;
  const int* xr = x + ((size_t)b * 2041 + t) * 9;
  for (int c = threadIdx.x; c < 896; c += blockDim.x) {
    float v;
    if (c < 64)        v = e0[xr[0] * 64  + c]         * 8.0f;
    else if (c < 96)   v = e1[xr[1] * 32  + (c - 64)]  * 5.65685424949238f;
    else if (c < 160)  v = e2[xr[2] * 64  + (c - 96)]  * 8.0f;
    else if (c < 672)  v = e3[xr[3] * 512 + (c - 160)] * 22.6274169979695f;
    else if (c < 800)  v = e4[xr[4] * 128 + (c - 672)] * 11.3137084989848f;
    else if (c < 832)  v = e5[xr[5] * 32  + (c - 800)] * 5.65685424949238f;
    else               v = e6[xr[6] * 64  + (c - 832)] * 8.0f;
    Er[c] = (_Float16)v;
  }
  for (int c = threadIdx.x; c < 256; c += blockDim.x)
    Tr[c] = (_Float16)(etime[xr[7] * 256 + c] * 16.0f);
}

// h += PE[pos] for main rows; also emit h_f16
__global__ __launch_bounds__(128) void pe_add_kernel(
    const int* __restrict__ x, float* __restrict__ h32, _Float16* __restrict__ h16)
{
  int r = blockIdx.x;
  int b = r >> 11, tt = r & 2047;
  if (tt < 7) return;
  int t = tt - 7;
  int pos = x[((size_t)b * 2041 + t) * 9 + 8];
  for (int c = threadIdx.x; c < 512; c += 128) {
    int i2 = c & ~1;
    float f = __expf((float)i2 * (-9.21034037197618f / 512.0f));
    float ang = (float)pos * f;
    float pe = (c & 1) ? __cosf(ang) : __sinf(ang);
    float v = h32[(size_t)r * 512 + c] + pe;
    h32[(size_t)r * 512 + c] = v;
    h16[(size_t)r * 512 + c] = (_Float16)v;
  }
}

// init rows: h[b*2048 + j] = init_embs(192) @ w_init_in + b_init_in, 28 rows total
__global__ __launch_bounds__(256) void init_rows_kernel(
    const int* __restrict__ init_token,
    const float* __restrict__ ie0, const float* __restrict__ ie1, const float* __restrict__ ie2,
    const float* __restrict__ w, const float* __restrict__ bias,
    float* __restrict__ h32, _Float16* __restrict__ h16)
{
  int r = blockIdx.x;           // 0..27
  int b = r / 7, j = r % 7;
  __shared__ float ie[192];
  if (threadIdx.x < 192) {
    int k = threadIdx.x;
    int seg = k >> 6, o = k & 63;
    int tok = init_token[(b * 7 + j) * 3 + seg];
    const float* tbl = (seg == 0) ? ie0 : ((seg == 1) ? ie1 : ie2);
    ie[k] = tbl[tok * 64 + o] * 8.0f;
  }
  __syncthreads();
  size_t row = (size_t)b * 2048 + j;
  for (int c = threadIdx.x; c < 512; c += blockDim.x) {
    float s = bias[c];
    for (int k = 0; k < 192; ++k) s += ie[k] * w[k * 512 + c];
    h32[row * 512 + c] = s;
    h16[row * 512 + c] = (_Float16)s;
  }
}

// LayerNorm over 512 (in-place on h32, also writes h16)
__global__ __launch_bounds__(128) void ln_kernel(
    float* __restrict__ h32, _Float16* __restrict__ h16,
    const float* __restrict__ g, const float* __restrict__ bb)
{
  __shared__ float red[128];
  const int r = blockIdx.x, tid = threadIdx.x;
  float* hr = h32 + (size_t)r * 512;
  float lv[4];
  float s = 0.0f;
#pragma unroll
  for (int i = 0; i < 4; ++i) { lv[i] = hr[tid + i * 128]; s += lv[i]; }
  red[tid] = s; __syncthreads();
  for (int st = 64; st > 0; st >>= 1) { if (tid < st) red[tid] += red[tid + st]; __syncthreads(); }
  float mean = red[0] * (1.0f / 512.0f);
  __syncthreads();
  s = 0.0f;
#pragma unroll
  for (int i = 0; i < 4; ++i) { float d = lv[i] - mean; s += d * d; }
  red[tid] = s; __syncthreads();
  for (int st = 64; st > 0; st >>= 1) { if (tid < st) red[tid] += red[tid + st]; __syncthreads(); }
  float inv = rsqrtf(red[0] * (1.0f / 512.0f) + 1e-5f);
#pragma unroll
  for (int i = 0; i < 4; ++i) {
    int c = tid + i * 128;
    float y = (lv[i] - mean) * inv * g[c] + bb[c];
    hr[c] = y;
    h16[(size_t)r * 512 + c] = (_Float16)y;
  }
}

// cat[8164 x 544] = [ h16 rows (skip init) | emb_1[y_type]*sqrt(32) ]
__global__ void cat_kernel(const _Float16* __restrict__ h16, const float* __restrict__ e1,
                           const int* __restrict__ y_type, _Float16* __restrict__ cat)
{
  const int total = 8164 * 544;
  for (int idx = blockIdx.x * blockDim.x + threadIdx.x; idx < total; idx += gridDim.x * blockDim.x) {
    int r = idx / 544, c = idx - r * 544;
    int b = r / 2041, t = r - b * 2041;
    _Float16 v;
    if (c < 512) v = h16[((size_t)b * 2048 + 7 + t) * 512 + c];
    else {
      int yt = y_type[b * 2041 + t];
      v = (_Float16)(e1[yt * 32 + (c - 512)] * 5.65685424949238f);
    }
    cat[(size_t)r * 544 + c] = v;
  }
}

// ---------------------------------------------------------------------------
extern "C" void kernel_launch(void* const* d_in, const int* in_sizes, int n_in,
                              void* d_out, int out_size, void* d_ws, size_t ws_size,
                              hipStream_t stream)
{
  (void)in_sizes; (void)n_in; (void)out_size; (void)ws_size;
  const float* ie0      = (const float*)d_in[0];
  const float* ie1      = (const float*)d_in[1];
  const float* ie2      = (const float*)d_in[2];
  const float* e0       = (const float*)d_in[3];
  const float* e1       = (const float*)d_in[4];
  const float* e2       = (const float*)d_in[5];
  const float* e3       = (const float*)d_in[6];
  const float* e4       = (const float*)d_in[7];
  const float* e5       = (const float*)d_in[8];
  const float* e6       = (const float*)d_in[9];
  const float* etime    = (const float*)d_in[10];
  const float* w_init   = (const float*)d_in[11];
  const float* b_init   = (const float*)d_in[12];
  const float* w_enc    = (const float*)d_in[13];
  const float* b_enc    = (const float*)d_in[14];
  const float* w_time   = (const float*)d_in[15];
  const float* b_time   = (const float*)d_in[16];
  const float* wq       = (const float*)d_in[17];
  const float* wk       = (const float*)d_in[18];
  const float* wv       = (const float*)d_in[19];
  const float* wo       = (const float*)d_in[20];
  const float* bq       = (const float*)d_in[21];
  const float* bk       = (const float*)d_in[22];
  const float* bv       = (const float*)d_in[23];
  const float* bo       = (const float*)d_in[24];
  const float* ln1_b    = (const float*)d_in[25];
  const float* ln2_b    = (const float*)d_in[26];
  const float* b_ff2    = (const float*)d_in[27];
  const float* ln1_g    = (const float*)d_in[28];
  const float* ln2_g    = (const float*)d_in[29];
  const float* w_ff1    = (const float*)d_in[30];
  const float* b_ff1    = (const float*)d_in[31];
  const float* w_ff2    = (const float*)d_in[32];
  const float* lnf_g    = (const float*)d_in[33];
  const float* lnf_b    = (const float*)d_in[34];
  const float* w_cat    = (const float*)d_in[35];
  const float* b_cat    = (const float*)d_in[36];
  const float* w_type   = (const float*)d_in[37];
  const float* b_type   = (const float*)d_in[38];
  const int*   x        = (const int*)d_in[51];
  const int*   init_tok = (const int*)d_in[52];
  const int*   y_type   = (const int*)d_in[53];
  float* out = (float*)d_out;

  char* ws = (char*)d_ws;
  size_t off = 0;
  auto alloc = [&](size_t bytes) -> char* {
    char* p = ws + off;
    off += (bytes + 255) & ~(size_t)255;
    return p;
  };
  const size_t MT = 8192;
  float*    h32   = (float*)   alloc(MT * 512 * 4);
  _Float16* h16   = (_Float16*)alloc(MT * 512 * 2);
  _Float16* E16   = (_Float16*)alloc(MT * 896 * 2);
  _Float16* T16   = (_Float16*)alloc(MT * 256 * 2);
  _Float16* q16   = (_Float16*)alloc(MT * 512 * 2);
  _Float16* k16   = (_Float16*)alloc(MT * 512 * 2);
  _Float16* v16b  = (_Float16*)alloc(MT * 512 * 2);
  _Float16* a16   = (_Float16*)alloc(MT * 512 * 2);
  _Float16* ff16  = (_Float16*)alloc(MT * 2048 * 2);
  _Float16* cat16 = (_Float16*)alloc(MT * 544 * 2);
  _Float16* y16   = (_Float16*)alloc(MT * 512 * 2);

  auto GEMM = [&](const _Float16* Ap, int lda, const float* Wp, const float* bp,
                  const float* rp, float* o32, _Float16* o16, int ldo, int nc0,
                  int M, int N, int K, int act) {
    dim3 g((N + 127) / 128, (M + 127) / 128);
    gemm_kernel<<<g, 256, 0, stream>>>(Ap, lda, Wp, bp, rp, o32, o16, ldo, nc0, M, N, K, act);
  };

  // -------- input encoding --------
  embed_gather_kernel<<<8192, 256, 0, stream>>>(x, e0, e1, e2, e3, e4, e5, e6, etime, E16, T16);
  GEMM(E16, 896, w_enc,  b_enc,  nullptr, h32, nullptr, 512, 0, 8192, 512, 896, 0);
  GEMM(T16, 256, w_time, b_time, h32,     h32, nullptr, 512, 0, 8192, 512, 256, 0);
  pe_add_kernel<<<8192, 128, 0, stream>>>(x, h32, h16);
  init_rows_kernel<<<28, 256, 0, stream>>>(init_tok, ie0, ie1, ie2, w_init, b_init, h32, h16);

  // -------- transformer layers --------
  for (int l = 0; l < 12; ++l) {
    const float* wql = wq + (size_t)l * 512 * 512;
    const float* wkl = wk + (size_t)l * 512 * 512;
    const float* wvl = wv + (size_t)l * 512 * 512;
    const float* wol = wo + (size_t)l * 512 * 512;
    GEMM(h16, 512, wql, bq + l * 512, nullptr, nullptr, q16,  512, 0, 8192, 512, 512, 2); // elu+1
    GEMM(h16, 512, wkl, bk + l * 512, nullptr, nullptr, k16,  512, 0, 8192, 512, 512, 2); // elu+1
    GEMM(h16, 512, wvl, bv + l * 512, nullptr, nullptr, v16b, 512, 0, 8192, 512, 512, 0);
    attn_kernel<<<32, 256, 0, stream>>>(q16, k16, v16b, a16);
    GEMM(a16, 512, wol, bo + l * 512, h32, h32, nullptr, 512, 0, 8192, 512, 512, 0);
    ln_kernel<<<8192, 128, 0, stream>>>(h32, h16, ln1_g + l * 512, ln1_b + l * 512);
    GEMM(h16,  512,  w_ff1 + (size_t)l * 512 * 2048, b_ff1 + l * 2048,
         nullptr, nullptr, ff16, 2048, 0, 8192, 2048, 512, 1);                            // gelu
    GEMM(ff16, 2048, w_ff2 + (size_t)l * 2048 * 512, b_ff2 + l * 512,
         h32, h32, nullptr, 512, 0, 8192, 512, 2048, 0);
    ln_kernel<<<8192, 128, 0, stream>>>(h32, h16, ln2_g + l * 512, ln2_b + l * 512);
  }
  ln_kernel<<<8192, 128, 0, stream>>>(h32, h16, lnf_g, lnf_b);

  // -------- output heads --------
  cat_kernel<<<2048, 256, 0, stream>>>(h16, e1, y_type, cat16);
  GEMM(cat16, 544, w_type, b_type, nullptr, out, nullptr, 278, 0, 8164, 5, 512, 0);
  GEMM(cat16, 544, w_cat,  b_cat,  nullptr, nullptr, y16, 512, 0, 8164, 512, 544, 0);
  const int head_w_idx[6] = {39, 41, 43, 45, 47, 49};
  const int head_n[6]     = {18, 65, 87, 20, 18, 65};
  const int head_off[6]   = {5, 23, 88, 175, 195, 213};
  for (int hN = 0; hN < 6; ++hN) {
    const float* wh = (const float*)d_in[head_w_idx[hN]];
    const float* bh = (const float*)d_in[head_w_idx[hN] + 1];
    GEMM(y16, 512, wh, bh, nullptr, out, nullptr, 278, head_off[hN], 8164, head_n[hN], 512, 0);
  }
}
